// GCN_65549790871804
// MI455X (gfx1250) — matrix operand order repaired and checked by reading, not compiled, not run on hardware
//
#include <hip/hip_runtime.h>
#include <math.h>

// GCN (2-layer, DGL GraphConv norm='both') for MI455X / gfx1250.
// Dense transforms use v_wmma_f32_16x16x32_f16 (wave32 WMMA, f32 accum).
// Edge aggregation uses L2-resident float atomics (agg buffers << 192MB L2).

typedef __attribute__((ext_vector_type(16))) _Float16 v16h;
typedef __attribute__((ext_vector_type(8)))  float    v8f;

#define IN_F  256
#define HID   32
#define OUT_F 40

// ---------------- utility kernels ----------------

__global__ void k_zero(float* __restrict__ p, long n) {
  long i = (long)blockIdx.x * blockDim.x + threadIdx.x;
  long stride = (long)gridDim.x * blockDim.x;
  for (; i < n; i += stride) p[i] = 0.0f;
}

__global__ void k_degree(const int* __restrict__ src, const int* __restrict__ dst,
                         float* __restrict__ deg_out, float* __restrict__ deg_in, int e) {
  int i = blockIdx.x * blockDim.x + threadIdx.x;
  if (i >= e) return;
  atomicAdd(&deg_out[src[i]], 1.0f);
  atomicAdd(&deg_in[dst[i]], 1.0f);
}

__global__ void k_invsqrt(float* __restrict__ deg_out, float* __restrict__ deg_in, int n) {
  int i = blockIdx.x * blockDim.x + threadIdx.x;
  if (i >= n) return;
  deg_out[i] = rsqrtf(fmaxf(deg_out[i], 1.0f));
  deg_in[i]  = rsqrtf(fmaxf(deg_in[i], 1.0f));
}

// ---------------- GEMM1: h = (x * inv_out) @ W1, [n x 256] @ [256 x 32] ----------------
// One wave per 16-node tile; 2 accumulators (N=0..15, N=16..31); K in steps of 32.

__global__ __launch_bounds__(256) void k_gemm1(const float* __restrict__ x,
                                               const float* __restrict__ W1,
                                               const float* __restrict__ inv_out,
                                               float* __restrict__ h, int n) {
  const int lane = threadIdx.x & 31;
  const int wave = threadIdx.x >> 5;
  const int tile = blockIdx.x * (blockDim.x >> 5) + wave;
  const int row0 = tile << 4;
  if (row0 >= n) return;                       // wave-uniform: EXEC stays all-1s
  const bool full = (row0 + 16 <= n);          // wave-uniform

  const int m    = lane & 15;
  const int half = lane >> 4;
  const int rowu = row0 + m;
  const int row  = (rowu < n) ? rowu : (n - 1);
  const float s  = (rowu < n) ? inv_out[row] : 0.0f;
  const float* xr = x + (long)row * IN_F;

  v8f c0 = {}; v8f c1 = {};
  for (int k0 = 0; k0 < IN_F; k0 += 32) {
    v16h a, b0, b1;
    // A-frag: lane holds row m; halfs j=0..15 -> K=k0+16h+j
    const float* pa = xr + k0 + 16 * half;
#pragma unroll
    for (int j = 0; j < 8; ++j) a[j] = (_Float16)(pa[j] * s);
    const float* pb = xr + k0 + 16 * half + 8;
#pragma unroll
    for (int j = 0; j < 8; ++j) a[8 + j] = (_Float16)(pb[j] * s);
    // B-frag: element j of lane = W1[k0 + j + 16h][N]; N = m (tile0) / m+16 (tile1)
    const float* w = W1 + (long)(k0 + 16 * half) * HID + m;
#pragma unroll
    for (int j = 0; j < 16; ++j) {
      b0[j] = (_Float16)w[j * HID];
      b1[j] = (_Float16)w[j * HID + 16];
    }
    c0 = __builtin_amdgcn_wmma_f32_16x16x32_f16(false, a, false, b0, (short)0, c0, false, false);
    c1 = __builtin_amdgcn_wmma_f32_16x16x32_f16(false, a, false, b1, (short)0, c1, false, false);
  }
  // D layout: VGPR r -> M = r + 8h, N = lane&15
  float* hr = h + (long)(row0 + 8 * half) * HID + m;
  if (full) {
    // n % 16 == 0 in this workload: branch-free coalesced epilogue
#pragma unroll
    for (int r = 0; r < 8; ++r) {
      hr[(long)r * HID]      = c0[r];
      hr[(long)r * HID + 16] = c1[r];
    }
  } else {
#pragma unroll
    for (int r = 0; r < 8; ++r) {
      if (row0 + r + 8 * half < n) {
        hr[(long)r * HID]      = c0[r];
        hr[(long)r * HID + 16] = c1[r];
      }
    }
  }
}

// ---------------- edge scatter, layer 1 (32 feats) ----------------
// One wave per edge (i>>5): wave-uniform src/dst loads, coalesced feature lanes,
// L2-resident global_atomic_add_f32.

__global__ void k_scatter32(const int* __restrict__ src, const int* __restrict__ dst,
                            const float* __restrict__ h, float* __restrict__ agg, long total) {
  long i = (long)blockIdx.x * blockDim.x + threadIdx.x;
  if (i >= total) return;
  int e = (int)(i >> 5);
  int f = (int)(i & 31);
  atomicAdd(&agg[(long)dst[e] * HID + f], h[(long)src[e] * HID + f]);
}

// ---------------- fused bias + ReLU + next-layer out-norm ----------------
// h1s = relu(agg1 * inv_in + b1) * inv_out   (written in place over h1)

__global__ void k_act1(const float* __restrict__ agg1, const float* __restrict__ inv_in,
                       const float* __restrict__ inv_out, const float* __restrict__ b1,
                       float* __restrict__ h1, int n) {
  long i = (long)blockIdx.x * blockDim.x + threadIdx.x;
  long total = (long)n * HID;
  if (i >= total) return;
  int node = (int)(i >> 5);
  int f    = (int)(i & 31);
  float v = fmaxf(agg1[i] * inv_in[node] + b1[f], 0.0f);
  h1[i] = v * inv_out[node];
}

// ---------------- GEMM2: h2 = h1s @ W2, [n x 32] @ [32 x 40] ----------------
// K=32 -> single WMMA step; N padded 40 -> 48 (3 tiles), pad via lane select.

__global__ __launch_bounds__(256) void k_gemm2(const float* __restrict__ h1,
                                               const float* __restrict__ W2,
                                               float* __restrict__ h2, int n) {
  const int lane = threadIdx.x & 31;
  const int wave = threadIdx.x >> 5;
  const int tile = blockIdx.x * (blockDim.x >> 5) + wave;
  const int row0 = tile << 4;
  if (row0 >= n) return;                       // wave-uniform
  const bool full = (row0 + 16 <= n);          // wave-uniform

  const int m    = lane & 15;
  const int half = lane >> 4;
  const int rowu = row0 + m;
  const int row  = (rowu < n) ? rowu : (n - 1);
  const float vs = (rowu < n) ? 1.0f : 0.0f;
  const float* hr = h1 + (long)row * HID;

  v16h a;
#pragma unroll
  for (int j = 0; j < 8; ++j) a[j]     = (_Float16)(hr[16 * half + j] * vs);
#pragma unroll
  for (int j = 0; j < 8; ++j) a[8 + j] = (_Float16)(hr[16 * half + 8 + j] * vs);

  v8f c[3] = {{}, {}, {}};
#pragma unroll
  for (int t = 0; t < 3; ++t) {
    const int nn = t * 16 + m;
    v16h b;
#pragma unroll
    for (int j = 0; j < 16; ++j) {
      int k = j + 16 * half;
      // lane-select padding keeps EXEC all-1s around the WMMAs
      b[j] = (nn < OUT_F) ? (_Float16)W2[k * OUT_F + nn] : (_Float16)0.0f;
    }
    c[t] = __builtin_amdgcn_wmma_f32_16x16x32_f16(false, a, false, b, (short)0, c[t], false, false);
  }

  float* o = h2 + (long)(row0 + 8 * half) * OUT_F;
  if (full) {
#pragma unroll
    for (int t = 0; t < 3; ++t) {
      const int nn = t * 16 + m;
      if (nn < OUT_F) {                         // single divergent guard per tile
#pragma unroll
        for (int r = 0; r < 8; ++r) o[(long)r * OUT_F + nn] = c[t][r];
      }
    }
  } else {
#pragma unroll
    for (int t = 0; t < 3; ++t) {
      const int nn = t * 16 + m;
      if (nn < OUT_F) {
#pragma unroll
        for (int r = 0; r < 8; ++r) {
          if (row0 + r + 8 * half < n) o[(long)r * OUT_F + nn] = c[t][r];
        }
      }
    }
  }
}

// ---------------- edge scatter, layer 2 (40 feats) ----------------

__global__ void k_scatter40(const int* __restrict__ src, const int* __restrict__ dst,
                            const float* __restrict__ h, float* __restrict__ agg, long total) {
  long i = (long)blockIdx.x * blockDim.x + threadIdx.x;
  if (i >= total) return;
  int e = (int)(i / OUT_F);
  int f = (int)(i % OUT_F);
  atomicAdd(&agg[(long)dst[e] * OUT_F + f], h[(long)src[e] * OUT_F + f]);
}

// ---------------- fused bias + log_softmax ----------------

__global__ void k_logsoftmax(const float* __restrict__ agg2, const float* __restrict__ inv_in,
                             const float* __restrict__ b2, float* __restrict__ out, int n) {
  int i = blockIdx.x * blockDim.x + threadIdx.x;
  if (i >= n) return;
  const float inv = inv_in[i];
  float z[OUT_F];
  float mx = -INFINITY;
#pragma unroll
  for (int c = 0; c < OUT_F; ++c) {
    z[c] = agg2[(long)i * OUT_F + c] * inv + b2[c];
    mx = fmaxf(mx, z[c]);
  }
  float sum = 0.0f;
#pragma unroll
  for (int c = 0; c < OUT_F; ++c) sum += expf(z[c] - mx);
  const float lse = mx + logf(sum);
#pragma unroll
  for (int c = 0; c < OUT_F; ++c) out[(long)i * OUT_F + c] = z[c] - lse;
}

// ---------------- launcher ----------------

extern "C" void kernel_launch(void* const* d_in, const int* in_sizes, int n_in,
                              void* d_out, int out_size, void* d_ws, size_t ws_size,
                              hipStream_t stream) {
  const float* x   = (const float*)d_in[0];
  const float* W1  = (const float*)d_in[1];
  const float* b1  = (const float*)d_in[2];
  const float* W2  = (const float*)d_in[3];
  const float* b2  = (const float*)d_in[4];
  const int*   src = (const int*)d_in[5];
  const int*   dst = (const int*)d_in[6];
  float* out = (float*)d_out;

  const int n = in_sizes[0] / IN_F;   // 100000
  const int e = in_sizes[5];          // 3200000

  // workspace layout (floats): deg/inv_out | deg/inv_in | h1 | agg1 | h2 | agg2
  float* ws      = (float*)d_ws;
  float* deg_out = ws;
  float* deg_in  = deg_out + n;
  float* h1      = deg_in + n;
  float* agg1    = h1   + (long)n * HID;
  float* h2      = agg1 + (long)n * HID;
  float* agg2    = h2   + (long)n * OUT_F;
  const long ws_total = (long)n * (2 + 2 * HID + 2 * OUT_F);

  k_zero<<<2048, 256, 0, stream>>>(ws, ws_total);

  k_degree<<<(e + 255) / 256, 256, 0, stream>>>(src, dst, deg_out, deg_in, e);
  k_invsqrt<<<(n + 255) / 256, 256, 0, stream>>>(deg_out, deg_in, n);

  const int tiles   = (n + 15) / 16;
  const int gblocks = (tiles + 7) / 8;       // 8 waves / block
  k_gemm1<<<gblocks, 256, 0, stream>>>(x, W1, deg_out, h1, n);

  const long t32 = (long)e * HID;
  k_scatter32<<<(int)((t32 + 255) / 256), 256, 0, stream>>>(src, dst, h1, agg1, t32);

  const long nh = (long)n * HID;
  k_act1<<<(int)((nh + 255) / 256), 256, 0, stream>>>(agg1, deg_in, deg_out, b1, h1, n);

  k_gemm2<<<gblocks, 256, 0, stream>>>(h1, W2, h2, n);

  const long t40 = (long)e * OUT_F;
  k_scatter40<<<(int)((t40 + 255) / 256), 256, 0, stream>>>(src, dst, h2, agg2, t40);

  k_logsoftmax<<<(n + 255) / 256, 256, 0, stream>>>(agg2, deg_in, b2, out, n);
}
